// LSTM_53171695124900
// MI455X (gfx1250) — matrix operand order repaired and checked by reading, not compiled, not run on hardware
//
#include <hip/hip_runtime.h>
#include <hip/hip_bf16.h>
#include <math.h>

// ---------------------------------------------------------------------------
// 2-layer LSTM (H=1024, T=4096, batch=1) for MI455X / gfx1250.
//
//  * k_convert : fp32 -> bf16 weights (W_hh1, W_ih2, W_hh2), fold b_ih2+b_hh2.
//  * k_xp      : Xp[t][4H] = x[t] @ W_ih1^T + b_ih1 + b_hh1 (parallel).
//  * k_lstm    : persistent kernel, 128 WGs x 256 threads. Each WG owns 8
//                hidden units; its 3x32 weight rows (192KB bf16) are DMA'd
//                into LDS once via TENSOR_LOAD_TO_LDS (12 descriptors/WG),
//                then 4096 steps of v_wmma_f32_16x16x32_bf16 mat-vecs.
//                2 grid barriers per step, h1/h2 double-buffered bf16,
//                c1/c2 stay in LDS.
//  * k_out     : reduce per-WG output partials -> out[T][2].
// ---------------------------------------------------------------------------

#define HSZ     1024
#define ISZ     64
#define OSZ     2
#define TSTEPS  4096
#define NWG     128
#define NTHR    256

typedef __attribute__((ext_vector_type(16))) __bf16 v16bf;
typedef __attribute__((ext_vector_type(8)))  float  v8f;
typedef __attribute__((ext_vector_type(4)))  unsigned u32x4;  // POD 16B vector
typedef __attribute__((ext_vector_type(8)))  int      i32x8;
typedef __attribute__((ext_vector_type(4)))  int      i32x4;

struct Frag32B { u32x4 lo, hi; };  // 32 bytes == 16 bf16

typedef unsigned short ushort_t;

__device__ __forceinline__ ushort_t f2bf(float f) {
  unsigned u = __builtin_bit_cast(unsigned, f);
  unsigned r = (u + 0x7FFFu + ((u >> 16) & 1u)) >> 16;  // RNE
  return (ushort_t)r;
}
__device__ __forceinline__ float sigm(float x) { return 1.0f / (1.0f + __expf(-x)); }

// ---------------------------------------------------------------------------
// Weight conversion + bias fold
// ---------------------------------------------------------------------------
__global__ void k_convert(const float* __restrict__ Whh1,
                          const float* __restrict__ Wih2,
                          const float* __restrict__ Whh2,
                          const float* __restrict__ bih2,
                          const float* __restrict__ bhh2,
                          ushort_t* __restrict__ Wb,     // [3][4H*H] bf16
                          float* __restrict__ bias2,     // [4H]
                          int n)                         // 4H*H
{
  for (int i = blockIdx.x * blockDim.x + threadIdx.x; i < n;
       i += gridDim.x * blockDim.x) {
    Wb[i]         = f2bf(Whh1[i]);
    Wb[n + i]     = f2bf(Wih2[i]);
    Wb[2 * n + i] = f2bf(Whh2[i]);
    if (i < 4 * HSZ) bias2[i] = bih2[i] + bhh2[i];
  }
}

// ---------------------------------------------------------------------------
// Xp[t][r] = b_ih1[r] + b_hh1[r] + sum_k x[t][k] * W_ih1[r][k]
// ---------------------------------------------------------------------------
__global__ void k_xp(const float* __restrict__ x,
                     const float* __restrict__ Wih1,
                     const float* __restrict__ bih1,
                     const float* __restrict__ bhh1,
                     float* __restrict__ Xp)
{
  int idx = blockIdx.x * blockDim.x + threadIdx.x;   // t*4H + r
  int t = idx >> 12, r = idx & 4095;
  const float* xr = x + t * ISZ;
  const float* wr = Wih1 + (size_t)r * ISZ;
  float acc = bih1[r] + bhh1[r];
#pragma unroll
  for (int k = 0; k < ISZ; ++k) acc = fmaf(xr[k], wr[k], acc);
  Xp[idx] = acc;
}

// ---------------------------------------------------------------------------
// Grid-wide barrier: monotonic counter, release/acquire at agent scope.
// ---------------------------------------------------------------------------
__device__ __forceinline__ void grid_barrier(unsigned* bar, unsigned target) {
  __threadfence();
  __syncthreads();
  if (threadIdx.x == 0) {
    __hip_atomic_fetch_add(bar, 1u, __ATOMIC_RELEASE, __HIP_MEMORY_SCOPE_AGENT);
    while (__hip_atomic_load(bar, __ATOMIC_ACQUIRE, __HIP_MEMORY_SCOPE_AGENT) < target)
      __builtin_amdgcn_s_sleep(2);
  }
  __syncthreads();
}

// ---------------------------------------------------------------------------
// One wave accumulates (16 gate rows) x (256-wide K slice) of W @ h into C
// via 8 x v_wmma_f32_16x16x32_bf16. A rows from LDS (lane-gathered), B
// columns broadcast h so every C column equals the mat-vec result.
// CDNA5 wave32 fragment layout (ISA 7.12.2):
//   A: lanes 0-15 row=lane,  elems = K {koff+0..7, koff+16..23}
//      lanes 16-31 row=l-16, elems = K {koff+8..15, koff+24..31}
//   B: lanes 0-15 elems = K {koff+0..7, koff+16..23} ; lanes 16-31 elems = K {koff+8..15, koff+24..31}
// ---------------------------------------------------------------------------
__device__ __forceinline__ void mv_accum(const ushort_t* __restrict__ wlds,
                                         int matBase,          // element offset
                                         int tile,             // 0 or 1
                                         const ushort_t* __restrict__ hglob,
                                         int kbase, v8f& c, int lane)
{
  const int row   = tile * 16 + (lane & 15);
  const int rbase = matBase + row * HSZ;
  const int asel  = (lane < 16) ? 0 : 8;
  const int hsel  = (lane < 16) ? 0 : 8;
#pragma unroll
  for (int ks = 0; ks < 8; ++ks) {
    const int koff = kbase + ks * 32;
    Frag32B fa, fb;
    fa.lo = *(const u32x4*)(wlds + rbase + koff + asel);
    fa.hi = *(const u32x4*)(wlds + rbase + koff + 16 + asel);
    const u32x4* hp = (const u32x4*)(hglob + koff + hsel);
    fb.lo = hp[0];
    fb.hi = hp[2];
    v16bf a = __builtin_bit_cast(v16bf, fa);
    v16bf b = __builtin_bit_cast(v16bf, fb);
    c = __builtin_amdgcn_wmma_f32_16x16x32_bf16(false, a, false, b,
                                                (short)0, c, false, false);
  }
}

// ---------------------------------------------------------------------------
// Persistent recurrent kernel.
// Dynamic LDS layout:
//   [0, 196608)            : 3 matrices x 32 rows x 1024 bf16 weights
//   part   (float[128])    : partial C columns, [tile][quarter][16]
//   c1,c2  (float[8] each) : cell states
//   h2s    (float[8])      : current-step h2 (for output partial)
// ---------------------------------------------------------------------------
__global__ void __launch_bounds__(NTHR, 1)
k_lstm(const float* __restrict__ Xp,        // [T][4H]
       const ushort_t* __restrict__ Wb,     // [3][4H*H] bf16
       const float* __restrict__ bias2,     // [4H]
       ushort_t* __restrict__ h1buf,        // [2][H] bf16 (double buffer)
       ushort_t* __restrict__ h2buf,        // [2][H] bf16
       const float* __restrict__ Wout,      // [2][H] fp32
       float* __restrict__ outP,            // [T][NWG][2]
       unsigned* __restrict__ bar)
{
  extern __shared__ char smem_raw[];
  ushort_t* wlds = (ushort_t*)smem_raw;
  float* part = (float*)(smem_raw + 3 * 32 * HSZ * 2);
  float* c1   = part + 128;
  float* c2   = part + 136;
  float* h2s  = part + 144;

  const int tid  = threadIdx.x;
  const int lane = tid & 31;
  const int wid  = tid >> 5;
  const int tile = wid >> 2;      // 0: gates i,f  1: gates g,o
  const int q    = wid & 3;       // K quarter (256 wide)
  const int w    = blockIdx.x;    // owns hidden units [w*8, w*8+8)

  const size_t mstride = (size_t)4 * HSZ * HSZ;

  // ---- one-time LDS weight fill: 12 TDM transfers of 16KB each.
  // Each (matrix m, gate g) block = 8 contiguous global rows x 1024 bf16
  // -> contiguous 16KB in both global and LDS (dynamic LDS base == 0 here,
  //    since this kernel declares no static __shared__ data).
#if __has_builtin(__builtin_amdgcn_tensor_load_to_lds)
  if (wid == 0) {
#pragma unroll
    for (int m = 0; m < 3; ++m) {
#pragma unroll
      for (int g = 0; g < 4; ++g) {
        const unsigned lds_off =
            (unsigned)((m * 32 + g * 8) * (HSZ * 2));          // bytes
        const unsigned long long ga = (unsigned long long)(uintptr_t)
            (Wb + m * mstride + ((size_t)g * HSZ + (size_t)w * 8) * HSZ);
        // D# group0: count=1 | lds_addr | global_addr[56:0] | type=2
        u32x4 g0 = { 1u,
                     lds_off,
                     (unsigned)(ga & 0xFFFFFFFFu),
                     (unsigned)((ga >> 32) & 0x1FFFFFFu) | (2u << 30) };
        // D# group1: data_size=1 (2B); tensor_dim0=tile_dim0=8192 elems;
        // tensor_dim1=tile_dim1=1; dim0_stride=8192; rest 0.
        i32x8 g1 = { (int)(1u << 16),          // data_size=2B
                     (int)(8192u << 16),       // tensor_dim0[15:0]<<16
                     (int)(1u << 16),          // dim0 hi=0 | tensor_dim1=1
                     (int)(8192u << 16),       // dim1 hi=0 | tile_dim0=8192
                     1,                        // tile_dim1=1 | tile_dim2=0
                     8192,                     // tensor_dim0_stride lo
                     0, 0 };
        i32x4 gz = { 0, 0, 0, 0 };
#if defined(__clang_major__) && __clang_major__ >= 23
        i32x8 gz8 = { 0, 0, 0, 0, 0, 0, 0, 0 };
        __builtin_amdgcn_tensor_load_to_lds(g0, g1, gz, gz, gz8, 0);
#else
        __builtin_amdgcn_tensor_load_to_lds(g0, g1, gz, gz, 0);
#endif
      }
    }
#if __has_builtin(__builtin_amdgcn_s_wait_tensorcnt)
    __builtin_amdgcn_s_wait_tensorcnt(0);
#else
    asm volatile("s_wait_tensorcnt 0x0" ::: "memory");
#endif
  }
#else  // fallback: manual gather copy (96 rows x 1024 bf16)
  for (int idx = tid; idx < 96 * 128; idx += NTHR) {
    int row = idx >> 7, chunk = idx & 127;
    int m = row >> 5, s = row & 31;
    int gr = (s >> 3) * HSZ + w * 8 + (s & 7);
    const u32x4* src = (const u32x4*)(Wb + m * mstride + (size_t)gr * HSZ) + chunk;
    u32x4* dst = (u32x4*)(wlds + (size_t)row * HSZ) + chunk;
    *dst = *src;
  }
#endif
  if (tid < 8) { c1[tid] = 0.0f; c2[tid] = 0.0f; }
  __syncthreads();

  for (int t = 0; t < TSTEPS; ++t) {
    const ushort_t* h1_prev = h1buf + ((t + 1) & 1) * HSZ;
    ushort_t*       h1_cur  = h1buf + (t & 1) * HSZ;
    const ushort_t* h2_prev = h2buf + ((t + 1) & 1) * HSZ;
    ushort_t*       h2_cur  = h2buf + (t & 1) * HSZ;

    if (t + 1 < TSTEPS)
      __builtin_prefetch(Xp + (size_t)(t + 1) * 4 * HSZ + tid * 16, 0, 1);

    // ---------------- layer 1: gates1 = Xp[t] + W_hh1 @ h1_prev ------------
    {
      v8f c = {0.f, 0.f, 0.f, 0.f, 0.f, 0.f, 0.f, 0.f};
      mv_accum(wlds, 0, tile, h1_prev, q * 256, c, lane);
      if ((lane & 15) == 0) {                 // lanes 0 & 16 hold column 0
        const int half = lane >> 4;           // 0: rows 0-7, 1: rows 8-15
        float* p = part + (tile * 4 + q) * 16 + half * 8;
#pragma unroll
        for (int r = 0; r < 8; ++r) p[r] = c[r];
      }
    }
    __syncthreads();
    if (tid < 8) {
      const int u = tid, j = w * 8 + u;
      float g4[4];
#pragma unroll
      for (int g = 0; g < 4; ++g) {
        int s = g * 8 + u, tl = s >> 4, r = s & 15;
        float acc = Xp[(size_t)t * 4 * HSZ + g * HSZ + j];
#pragma unroll
        for (int qq = 0; qq < 4; ++qq) acc += part[(tl * 4 + qq) * 16 + r];
        g4[g] = acc;
      }
      float ig = sigm(g4[0]), fg = sigm(g4[1]);
      float gg = tanhf(g4[2]), og = sigm(g4[3]);
      float cn = fg * c1[u] + ig * gg;
      c1[u] = cn;
      h1_cur[j] = f2bf(og * tanhf(cn));
    }
    grid_barrier(bar, (unsigned)(2 * t + 1) * NWG);

    // ---------- layer 2: gates2 = bias2 + W_ih2 @ h1_cur + W_hh2 @ h2_prev -
    {
      v8f c = {0.f, 0.f, 0.f, 0.f, 0.f, 0.f, 0.f, 0.f};
      mv_accum(wlds, 32 * HSZ, tile, h1_cur,  q * 256, c, lane);
      mv_accum(wlds, 64 * HSZ, tile, h2_prev, q * 256, c, lane);
      if ((lane & 15) == 0) {
        const int half = lane >> 4;
        float* p = part + (tile * 4 + q) * 16 + half * 8;
#pragma unroll
        for (int r = 0; r < 8; ++r) p[r] = c[r];
      }
    }
    __syncthreads();
    if (tid < 8) {
      const int u = tid, j = w * 8 + u;
      float g4[4];
#pragma unroll
      for (int g = 0; g < 4; ++g) {
        int s = g * 8 + u, tl = s >> 4, r = s & 15;
        float acc = bias2[g * HSZ + j];
#pragma unroll
        for (int qq = 0; qq < 4; ++qq) acc += part[(tl * 4 + qq) * 16 + r];
        g4[g] = acc;
      }
      float ig = sigm(g4[0]), fg = sigm(g4[1]);
      float gg = tanhf(g4[2]), og = sigm(g4[3]);
      float cn = fg * c2[u] + ig * gg;
      c2[u] = cn;
      float hn = og * tanhf(cn);
      h2s[u] = hn;
      h2_cur[j] = f2bf(hn);
    }
    grid_barrier(bar, (unsigned)(2 * t + 2) * NWG);

    // ---- per-WG output partials: outP[t][w][o] = sum_u h2[u]*Wout[o][j]
    if (tid < OSZ) {
      float acc = 0.0f;
#pragma unroll
      for (int u = 0; u < 8; ++u)
        acc = fmaf(h2s[u], Wout[tid * HSZ + w * 8 + u], acc);
      outP[((size_t)t * NWG + w) * OSZ + tid] = acc;
    }
  }
}

// ---------------------------------------------------------------------------
// Final reduction: out[t][o] = b_out[o] + sum_w outP[t][w][o]
// ---------------------------------------------------------------------------
__global__ void k_out(const float* __restrict__ outP,
                      const float* __restrict__ bout,
                      float* __restrict__ out)
{
  int idx = blockIdx.x * blockDim.x + threadIdx.x;  // t*2 + o
  if (idx >= TSTEPS * OSZ) return;
  int t = idx >> 1, o = idx & 1;
  float acc = bout[o];
  for (int w = 0; w < NWG; ++w) acc += outP[((size_t)t * NWG + w) * OSZ + o];
  out[idx] = acc;
}

// ---------------------------------------------------------------------------
extern "C" void kernel_launch(void* const* d_in, const int* in_sizes, int n_in,
                              void* d_out, int out_size, void* d_ws, size_t ws_size,
                              hipStream_t stream)
{
  const float* x    = (const float*)d_in[0];
  const float* Wih1 = (const float*)d_in[1];
  const float* Whh1 = (const float*)d_in[2];
  const float* bih1 = (const float*)d_in[3];
  const float* bhh1 = (const float*)d_in[4];
  const float* Wih2 = (const float*)d_in[5];
  const float* Whh2 = (const float*)d_in[6];
  const float* bih2 = (const float*)d_in[7];
  const float* bhh2 = (const float*)d_in[8];
  const float* Wout = (const float*)d_in[9];
  const float* bout = (const float*)d_in[10];

  // Workspace layout (256B aligned regions):
  char* ws = (char*)d_ws;
  unsigned* bar   = (unsigned*)ws;                       // [0, 256)
  ushort_t* h1buf = (ushort_t*)(ws + 256);               // 2*1024*2  = 4096
  ushort_t* h2buf = (ushort_t*)(ws + 256 + 4096);        // 4096
  float*    bias2 = (float*)(ws + 8448);                 // 4H*4      = 16384
  float*    outP  = (float*)(ws + 24832);                // T*NWG*2*4 = 4 MiB
  ushort_t* Wb    = (ushort_t*)(ws + 4219136);           // 3*4H*H*2  = 24 MiB
  float*    Xp    = (float*)(ws + 29384960);             // T*4H*4    = 64 MiB

  // Reset barrier counter + zero initial h states (deterministic per launch).
  hipMemsetAsync(ws, 0, 8448, stream);

  const int nW = 4 * HSZ * HSZ;
  k_convert<<<4096, NTHR, 0, stream>>>(Whh1, Wih2, Whh2, bih2, bhh2, Wb, bias2, nW);
  k_xp<<<(TSTEPS * 4 * HSZ) / NTHR, NTHR, 0, stream>>>(x, Wih1, bih1, bhh1, Xp);

  const size_t lds_bytes = 3 * 32 * HSZ * 2 + 1024;      // 197632 B
  k_lstm<<<NWG, NTHR, lds_bytes, stream>>>(Xp, Wb, bias2, h1buf, h2buf,
                                           Wout, outP, bar);

  k_out<<<(TSTEPS * OSZ + NTHR - 1) / NTHR, NTHR, 0, stream>>>(outP, bout,
                                                               (float*)d_out);
}
